// EnhancedMultiHeadAttention_67783173865510
// MI455X (gfx1250) — matrix operand order repaired and checked by reading, not compiled, not run on hardware
//
#include <hip/hip_runtime.h>
#include <cstdint>
#include <cstddef>

// ---------------- constants ----------------
#define D_MODEL   1024
#define N_HEADS   16
#define D_K       64
#define SEQ       1024
#define BATCH     4
#define M_TOT     (BATCH * SEQ)   // 4096 rows
#define RANK      16
#define LORA_SCALE 2.0f
#define LN_EPS    1e-5f

// ---------------- types ----------------
typedef __attribute__((ext_vector_type(16))) __bf16 v16bf;
typedef __attribute__((ext_vector_type(8)))  float  v8f;

union Frag16 { v16bf v; uint4 q[2]; };

__device__ __forceinline__ v8f v8f_zero() {
  v8f z = {0.f, 0.f, 0.f, 0.f, 0.f, 0.f, 0.f, 0.f};
  return z;
}

__device__ __forceinline__ unsigned short f2bf(float f) {
  unsigned u = __float_as_uint(f);
  u += 0x7FFFu + ((u >> 16) & 1u);            // round-to-nearest-even
  return (unsigned short)(u >> 16);
}
__device__ __forceinline__ float bf2f(unsigned short h) {
  return __uint_as_float(((unsigned)h) << 16);
}

__device__ __forceinline__ v8f wmma_bf16(const v16bf& a, const v16bf& b, const v8f& c) {
  return __builtin_amdgcn_wmma_f32_16x16x32_bf16(false, a, false, b, (short)0, c, false, false);
}

// ---- CDNA5 async global -> LDS copy, SADDR (GVS) form ----
// mem_addr = SGPR64 + VGPR32(byte offset) + IOFFSET ; LDS addr = VDST + IOFFSET.
__device__ __forceinline__ void wait_asynccnt_0() {
  asm volatile("s_wait_asynccnt 0" ::: "memory");
}
__device__ __forceinline__ void wait_asynccnt_5() {
  asm volatile("s_wait_asynccnt 5" ::: "memory");
}

// ---------------- f32 -> bf16 convert ----------------
__global__ void k_f32_to_bf16(const float* __restrict__ src,
                              unsigned short* __restrict__ dst, int n) {
  int i = blockIdx.x * blockDim.x + threadIdx.x;
  if (i < n) dst[i] = f2bf(src[i]);
}

// ---------------- rank-16 LoRA pre-product: T = X @ A  ----------------
__global__ void k_xA_f32(const float* __restrict__ X, const float* __restrict__ A,
                         float* __restrict__ T) {
  int idx = blockIdx.x * blockDim.x + threadIdx.x;   // M_TOT*RANK threads
  int m = idx >> 4, r = idx & 15;
  const float* xr = X + (size_t)m * D_MODEL;
  float acc = 0.f;
  for (int k = 0; k < D_MODEL; ++k) acc += xr[k] * A[k * RANK + r];
  T[idx] = acc;
}

__global__ void k_xA_bf16(const unsigned short* __restrict__ X, const float* __restrict__ A,
                          float* __restrict__ T) {
  int idx = blockIdx.x * blockDim.x + threadIdx.x;
  int m = idx >> 4, r = idx & 15;
  const unsigned short* xr = X + (size_t)m * D_MODEL;
  float acc = 0.f;
  for (int k = 0; k < D_MODEL; ++k) acc += bf2f(xr[k]) * A[k * RANK + r];
  T[idx] = acc;
}

// ---------------- WMMA GEMM: Y0[M][1024] = Xb[M][1024] @ Wb[n][k]^T ----------------
// Block = 256 threads (8 waves). Wave tile 32(M) x 64(N); block tile 64 x 256.
// A (64x32) and B (256x32) K-tiles double-buffered in LDS, filled with
// global_load_async_to_lds_b128 (ASYNCcnt, SADDR form), consumed via ds_load_b128.
// K-pipeline manually unrolled x2 so buffer addresses are loop-invariant.
__global__ __launch_bounds__(256, 1) void k_gemm_bf16(
    const unsigned short* __restrict__ Xb,   // [M][1024] bf16 row-major
    const unsigned short* __restrict__ Wb,   // [1024 n][1024 k] bf16 (orig torch layout)
    float* __restrict__ Y0) {
  const int lane = threadIdx.x & 31;
  const int w    = threadIdx.x >> 5;
  const int m0b  = blockIdx.x * 64;
  const int n0b  = blockIdx.y * 256;
  const int wm   = (w >> 2) * 32;    // wave M offset in block tile
  const int wn   = (w & 3) * 64;     // wave N offset in block tile
  const int lx   = lane & 15;
  const int hi   = lane >> 4;        // lane half
  const int kbA  = hi * 16;          // A frag K base offset (contiguous k = 16*hi + s)
  const int kbB  = hi * 16;          // B frag K base offset

  __shared__ __align__(16) unsigned short ldsA[2][64 * 32];   //  8 KB
  __shared__ __align__(16) unsigned short ldsB[2][256 * 32];  // 32 KB

  const int t = threadIdx.x;
  // A: one 16B chunk / thread: tile row t>>2, cols (t&3)*8 ..+7
  const unsigned gofsA  = (unsigned)(((m0b + (t >> 2)) * 1024 + (t & 3) * 8) * 2);
  // B: one contiguous 64B tile row / thread (4 x b128 with ioffset 0/16/32/48;
  //    ioffset advances LDS and global addresses by the same delta)
  const unsigned gofsB  = (unsigned)((n0b + t) * 1024 * 2);
  const unsigned lofsA0 = (unsigned)(size_t)&ldsA[0][(t >> 2) * 32 + (t & 3) * 8];
  const unsigned lofsA1 = lofsA0 + 64 * 32 * 2;
  const unsigned lofsB0 = (unsigned)(size_t)&ldsB[0][t * 32];
  const unsigned lofsB1 = lofsB0 + 256 * 32 * 2;

  v8f acc[8];
  #pragma unroll
  for (int i = 0; i < 8; ++i) acc[i] = v8f_zero();

  auto stage = [&](unsigned la, unsigned lb, int k) {
    const unsigned short* sa = Xb + k;   // uniform SADDR base advance
    const unsigned short* sb = Wb + k;
    asm volatile("global_load_async_to_lds_b128 %0, %1, %2"
                 :: "v"(la), "v"(gofsA), "s"(sa) : "memory");
    asm volatile("global_load_async_to_lds_b128 %0, %1, %2"
                 :: "v"(lb), "v"(gofsB), "s"(sb) : "memory");
    asm volatile("global_load_async_to_lds_b128 %0, %1, %2 offset:16"
                 :: "v"(lb), "v"(gofsB), "s"(sb) : "memory");
    asm volatile("global_load_async_to_lds_b128 %0, %1, %2 offset:32"
                 :: "v"(lb), "v"(gofsB), "s"(sb) : "memory");
    asm volatile("global_load_async_to_lds_b128 %0, %1, %2 offset:48"
                 :: "v"(lb), "v"(gofsB), "s"(sb) : "memory");
  };

  auto compute = [&](const unsigned short* bufA, const unsigned short* bufB) {
    Frag16 a[2], b[4];
    #pragma unroll
    for (int i = 0; i < 2; ++i) {
      const unsigned short* p = bufA + (wm + i * 16 + lx) * 32 + kbA;
      a[i].q[0] = *(const uint4*)p;
      a[i].q[1] = *(const uint4*)(p + 8);
    }
    #pragma unroll
    for (int j = 0; j < 4; ++j) {
      const unsigned short* p = bufB + (wn + j * 16 + lx) * 32 + kbB;
      b[j].q[0] = *(const uint4*)p;
      b[j].q[1] = *(const uint4*)(p + 8);
    }
    #pragma unroll
    for (int i = 0; i < 2; ++i)
      #pragma unroll
      for (int j = 0; j < 4; ++j)
        acc[i * 4 + j] = wmma_bf16(a[i].v, b[j].v, acc[i * 4 + j]);
  };

  // ---- software pipeline: stage tile 0, then 16 x (two phases of K=32) ----
  stage(lofsA0, lofsB0, 0);
  for (int kp = 0; kp < 16; ++kp) {
    const int k0 = kp * 64;
    // phase 0: compute buffer 0 (K = k0), prefetch buffer 1 (K = k0+32)
    stage(lofsA1, lofsB1, k0 + 32);
    wait_asynccnt_5();                 // buffer-0 stage complete
    __syncthreads();
    compute(&ldsA[0][0], &ldsB[0][0]);
    __syncthreads();                   // all reads of buf0 done before restage
    // phase 1: compute buffer 1 (K = k0+32), prefetch buffer 0 (K = k0+64)
    if (kp + 1 < 16) {
      stage(lofsA0, lofsB0, k0 + 64);
      wait_asynccnt_5();               // buffer-1 stage complete
    } else {
      wait_asynccnt_0();
    }
    __syncthreads();
    compute(&ldsA[1][0], &ldsB[1][0]);
    __syncthreads();
  }

  #pragma unroll
  for (int i = 0; i < 2; ++i)
    #pragma unroll
    for (int j = 0; j < 4; ++j) {
      int n = n0b + wn + j * 16 + lx;
      #pragma unroll
      for (int v = 0; v < 8; ++v) {
        int m = m0b + wm + i * 16 + v + 8 * hi;
        Y0[(size_t)m * D_MODEL + n] = acc[i * 4 + j][v];
      }
    }
}

// ---------------- QKV epilogue: +bias +LoRA, scatter to head layouts (bf16) ------------
__global__ void k_epilogue_qkv(const float* __restrict__ Y0, const float* __restrict__ bias,
                               const float* __restrict__ T, const float* __restrict__ Blora,
                               unsigned short* __restrict__ dst, int isVT) {
  int idx = blockIdx.x * blockDim.x + threadIdx.x;   // M_TOT*D_MODEL
  int m = idx >> 10, n = idx & 1023;
  float y = Y0[idx] + bias[n];
  const float* t = T + (size_t)m * RANK;
  float l = 0.f;
  #pragma unroll
  for (int r = 0; r < RANK; ++r) l += t[r] * Blora[r * D_MODEL + n];
  y += LORA_SCALE * l;
  int b = m >> 10, s = m & 1023, h = n >> 6, d = n & 63;
  size_t o;
  if (isVT) o = ((size_t)(b * N_HEADS + h) * D_K + d) * SEQ + s;   // V^T: [b][h][d][s]
  else      o = ((size_t)(b * N_HEADS + h) * SEQ + s) * D_K + d;   // Q/K: [b][h][s][d]
  dst[o] = f2bf(y);
}

// ---------------- O epilogue: +bias +LoRA +residual (in-place f32) ----------------
__global__ void k_epilogue_o(float* __restrict__ Y0, const float* __restrict__ bias,
                             const float* __restrict__ T, const float* __restrict__ Blora,
                             const float* __restrict__ Xres) {
  int idx = blockIdx.x * blockDim.x + threadIdx.x;
  int m = idx >> 10, n = idx & 1023;
  const float* t = T + (size_t)m * RANK;
  float l = 0.f;
  #pragma unroll
  for (int r = 0; r < RANK; ++r) l += t[r] * Blora[r * D_MODEL + n];
  Y0[idx] = Y0[idx] + bias[n] + LORA_SCALE * l + Xres[idx];
}

// ---------------- flash attention: ctx + per-row softmax stats ----------------
// Grid: (SEQ/64, N_HEADS, BATCH). Block 128 (4 waves); wave owns 16 q rows.
__global__ __launch_bounds__(128) void k_attn_flash(
    const unsigned short* __restrict__ Qb,   // [b][h][s][d] bf16
    const unsigned short* __restrict__ Kb,   // [b][h][s][d] bf16
    const unsigned short* __restrict__ VT,   // [b][h][d][s] bf16
    unsigned short* __restrict__ ctxb,       // [b][s][1024] bf16
    float* __restrict__ rowmax_o, float* __restrict__ rowsum_o) {
  const int lane  = threadIdx.x & 31;
  const int w     = threadIdx.x >> 5;       // 0..3
  const int b     = blockIdx.z, h = blockIdx.y;
  const int qbase = blockIdx.x * 64 + w * 16;
  const int lx    = lane & 15;
  const int hi    = lane >> 4;
  const int kbA   = hi * 16;
  const int kbB   = hi * 16;

  const unsigned short* Qh = Qb + (size_t)(b * N_HEADS + h) * SEQ * D_K;
  const unsigned short* Kh = Kb + (size_t)(b * N_HEADS + h) * SEQ * D_K;
  const unsigned short* Vh = VT + (size_t)(b * N_HEADS + h) * D_K * SEQ;

  __shared__ __align__(16) unsigned short probs[4][16 * 32];

  // preload Q A-fragments (K dim = d, 2 steps of 32)
  Frag16 aq[2];
  #pragma unroll
  for (int i = 0; i < 2; ++i) {
    const unsigned short* p = Qh + (size_t)(qbase + lx) * D_K + i * 32 + kbA;
    aq[i].q[0] = *(const uint4*)p;
    aq[i].q[1] = *(const uint4*)(p + 8);
  }

  v8f acc[4];
  #pragma unroll
  for (int j = 0; j < 4; ++j) acc[j] = v8f_zero();
  float rmax[8], rsum[8];
  #pragma unroll
  for (int v = 0; v < 8; ++v) { rmax[v] = -1e30f; rsum[v] = 0.f; }

  for (int sk = 0; sk < SEQ; sk += 32) {
    // ---- scores for two 16-col tiles ----
    v8f sc[2];
    #pragma unroll
    for (int t = 0; t < 2; ++t) {
      Frag16 bk0, bk1;
      const unsigned short* p = Kh + (size_t)(sk + t * 16 + lx) * D_K;
      bk0.q[0] = *(const uint4*)(p + kbB);
      bk0.q[1] = *(const uint4*)(p + kbB + 8);
      bk1.q[0] = *(const uint4*)(p + 32 + kbB);
      bk1.q[1] = *(const uint4*)(p + 32 + kbB + 8);
      v8f z = v8f_zero();
      sc[t] = wmma_bf16(aq[0].v, bk0.v, z);
      sc[t] = wmma_bf16(aq[1].v, bk1.v, sc[t]);
      #pragma unroll
      for (int v = 0; v < 8; ++v) sc[t][v] *= 0.125f;   // 1/sqrt(64)
    }
    // ---- row max (cols live in low 4 lane bits of each half) ----
    float tmax[8];
    #pragma unroll
    for (int v = 0; v < 8; ++v) tmax[v] = fmaxf(sc[0][v], sc[1][v]);
    #pragma unroll
    for (int off = 1; off <= 8; off <<= 1)
      #pragma unroll
      for (int v = 0; v < 8; ++v) tmax[v] = fmaxf(tmax[v], __shfl_xor(tmax[v], off, 32));
    float fac[8], tsum[8];
    #pragma unroll
    for (int v = 0; v < 8; ++v) {
      float nm = fmaxf(rmax[v], tmax[v]);
      fac[v] = __expf(rmax[v] - nm);
      rmax[v] = nm;
    }
    #pragma unroll
    for (int v = 0; v < 8; ++v) {
      sc[0][v] = __expf(sc[0][v] - rmax[v]);
      sc[1][v] = __expf(sc[1][v] - rmax[v]);
      tsum[v]  = sc[0][v] + sc[1][v];
    }
    #pragma unroll
    for (int off = 1; off <= 8; off <<= 1)
      #pragma unroll
      for (int v = 0; v < 8; ++v) tsum[v] += __shfl_xor(tsum[v], off, 32);
    #pragma unroll
    for (int v = 0; v < 8; ++v) rsum[v] = rsum[v] * fac[v] + tsum[v];
    #pragma unroll
    for (int j = 0; j < 4; ++j)
      #pragma unroll
      for (int v = 0; v < 8; ++v) acc[j][v] *= fac[v];

    // ---- C-layout -> A-layout repack through LDS (bf16) ----
    #pragma unroll
    for (int t = 0; t < 2; ++t)
      #pragma unroll
      for (int v = 0; v < 8; ++v)
        probs[w][(v + 8 * hi) * 32 + t * 16 + lx] = f2bf(sc[t][v]);
    __syncthreads();
    Frag16 ap;
    {
      const unsigned short* p = &probs[w][lx * 32 + kbA];
      ap.q[0] = *(const uint4*)p;
      ap.q[1] = *(const uint4*)(p + 8);
    }
    // ---- ctx += probs @ V  (B from V^T, contiguous along s) ----
    #pragma unroll
    for (int j = 0; j < 4; ++j) {
      Frag16 bv;
      const unsigned short* p = Vh + (size_t)(j * 16 + lx) * SEQ + sk + kbB;
      bv.q[0] = *(const uint4*)p;
      bv.q[1] = *(const uint4*)(p + 8);
      acc[j] = wmma_bf16(ap.v, bv.v, acc[j]);
    }
    __syncthreads();
  }

  // ---- stats out, then normalized ctx (bf16, [b][s][h*64+d]) ----
  if (lx == 0) {
    #pragma unroll
    for (int v = 0; v < 8; ++v) {
      int s = qbase + v + 8 * hi;
      size_t o = (size_t)(b * N_HEADS + h) * SEQ + s;
      rowmax_o[o] = rmax[v];
      rowsum_o[o] = rsum[v];
    }
  }
  float inv[8];
  #pragma unroll
  for (int v = 0; v < 8; ++v) inv[v] = 1.f / rsum[v];
  #pragma unroll
  for (int j = 0; j < 4; ++j)
    #pragma unroll
    for (int v = 0; v < 8; ++v) {
      int s = qbase + v + 8 * hi;
      int d = j * 16 + lx;
      ctxb[(size_t)(b * SEQ + s) * D_MODEL + h * D_K + d] = f2bf(acc[j][v] * inv[v]);
    }
}

// ---------------- attn.mean over heads (pass 2, WMMA recompute) ----------------
// Grid: (SEQ/16, SEQ/128, BATCH). Block 256; wave owns a 16x16 (q,k) tile.
__global__ __launch_bounds__(256) void k_attn_mean(
    const unsigned short* __restrict__ Qb, const unsigned short* __restrict__ Kb,
    const float* __restrict__ rowmax_i, const float* __restrict__ rowsum_i,
    float* __restrict__ out_attn) {
  const int lane  = threadIdx.x & 31;
  const int w     = threadIdx.x >> 5;
  const int b     = blockIdx.z;
  const int qbase = blockIdx.x * 16;
  const int ktile = blockIdx.y * 128 + w * 16;
  const int lx    = lane & 15;
  const int hi    = lane >> 4;
  const int kbA   = hi * 16;
  const int kbB   = hi * 16;

  v8f acc = v8f_zero();
  for (int h = 0; h < N_HEADS; ++h) {
    const unsigned short* Qh = Qb + (size_t)(b * N_HEADS + h) * SEQ * D_K;
    const unsigned short* Kh = Kb + (size_t)(b * N_HEADS + h) * SEQ * D_K;
    Frag16 aq0, aq1, bk0, bk1;
    const unsigned short* pq = Qh + (size_t)(qbase + lx) * D_K;
    aq0.q[0] = *(const uint4*)(pq + kbA);
    aq0.q[1] = *(const uint4*)(pq + kbA + 8);
    aq1.q[0] = *(const uint4*)(pq + 32 + kbA);
    aq1.q[1] = *(const uint4*)(pq + 32 + kbA + 8);
    const unsigned short* pk = Kh + (size_t)(ktile + lx) * D_K;
    bk0.q[0] = *(const uint4*)(pk + kbB);
    bk0.q[1] = *(const uint4*)(pk + kbB + 8);
    bk1.q[0] = *(const uint4*)(pk + 32 + kbB);
    bk1.q[1] = *(const uint4*)(pk + 32 + kbB + 8);
    v8f sc = v8f_zero();
    sc = wmma_bf16(aq0.v, bk0.v, sc);
    sc = wmma_bf16(aq1.v, bk1.v, sc);
    const float* mx = rowmax_i + (size_t)(b * N_HEADS + h) * SEQ;
    const float* sm = rowsum_i + (size_t)(b * N_HEADS + h) * SEQ;
    #pragma unroll
    for (int v = 0; v < 8; ++v) {
      int q = qbase + v + 8 * hi;
      acc[v] += __expf(sc[v] * 0.125f - mx[q]) * (1.f / sm[q]);
    }
  }
  #pragma unroll
  for (int v = 0; v < 8; ++v) {
    int q = qbase + v + 8 * hi;
    out_attn[(size_t)(b * SEQ + q) * SEQ + ktile + lx] = acc[v] * (1.f / N_HEADS);
  }
}

// ---------------- LayerNorm (block per row) ----------------
__global__ __launch_bounds__(256) void k_layernorm(
    const float* __restrict__ H, const float* __restrict__ g,
    const float* __restrict__ bb, float* __restrict__ out) {
  int m = blockIdx.x;
  const float* row = H + (size_t)m * D_MODEL;
  float x[4], s = 0.f, ss = 0.f;
  #pragma unroll
  for (int i = 0; i < 4; ++i) {
    x[i] = row[threadIdx.x + i * 256];
    s += x[i]; ss += x[i] * x[i];
  }
  __shared__ float rs[256], rss[256];
  rs[threadIdx.x] = s; rss[threadIdx.x] = ss;
  __syncthreads();
  for (int off = 128; off > 0; off >>= 1) {
    if (threadIdx.x < off) {
      rs[threadIdx.x]  += rs[threadIdx.x + off];
      rss[threadIdx.x] += rss[threadIdx.x + off];
    }
    __syncthreads();
  }
  float mean = rs[0] * (1.f / D_MODEL);
  float var  = rss[0] * (1.f / D_MODEL) - mean * mean;
  float inv  = rsqrtf(var + LN_EPS);
  #pragma unroll
  for (int i = 0; i < 4; ++i) {
    int n = threadIdx.x + i * 256;
    out[(size_t)m * D_MODEL + n] = (x[i] - mean) * inv * g[n] + bb[n];
  }
}

// ---------------- host orchestration ----------------
extern "C" void kernel_launch(void* const* d_in, const int* in_sizes, int n_in,
                              void* d_out, int out_size, void* d_ws, size_t ws_size,
                              hipStream_t stream) {
  (void)in_sizes; (void)n_in; (void)out_size; (void)ws_size;
  const float* x = (const float*)d_in[0];
  const float* wmat[4] = {(const float*)d_in[1], (const float*)d_in[5],
                          (const float*)d_in[9], (const float*)d_in[13]};
  const float* bias[4] = {(const float*)d_in[2], (const float*)d_in[6],
                          (const float*)d_in[10], (const float*)d_in[14]};
  const float* Amat[4] = {(const float*)d_in[3], (const float*)d_in[7],
                          (const float*)d_in[11], (const float*)d_in[15]};
  const float* Bmat[4] = {(const float*)d_in[4], (const float*)d_in[8],
                          (const float*)d_in[12], (const float*)d_in[16]};
  const float* ln_g = (const float*)d_in[17];
  const float* ln_b = (const float*)d_in[18];
  float* out = (float*)d_out;

  char* ws = (char*)d_ws;
  size_t off = 0;
  auto alloc = [&](size_t bytes) -> char* {
    char* p = ws + off;
    off += (bytes + 255) & ~(size_t)255;
    return p;
  };
  unsigned short* xb = (unsigned short*)alloc((size_t)M_TOT * D_MODEL * 2);
  unsigned short* wb[4];
  for (int p = 0; p < 4; ++p) wb[p] = (unsigned short*)alloc((size_t)D_MODEL * D_MODEL * 2);
  float* T[4];
  for (int p = 0; p < 4; ++p) T[p] = (float*)alloc((size_t)M_TOT * RANK * 4);
  float* Y0 = (float*)alloc((size_t)M_TOT * D_MODEL * 4);
  unsigned short* Qbuf = (unsigned short*)alloc((size_t)M_TOT * D_MODEL * 2);
  unsigned short* Kbuf = (unsigned short*)alloc((size_t)M_TOT * D_MODEL * 2);
  unsigned short* Vtb  = (unsigned short*)alloc((size_t)M_TOT * D_MODEL * 2);
  float* rmax = (float*)alloc((size_t)BATCH * N_HEADS * SEQ * 4);
  float* rsum = (float*)alloc((size_t)BATCH * N_HEADS * SEQ * 4);
  unsigned short* ctxb = (unsigned short*)alloc((size_t)M_TOT * D_MODEL * 2);

  const int NELEM = M_TOT * D_MODEL;          // 4M
  // 1) bf16 conversions
  k_f32_to_bf16<<<NELEM / 256, 256, 0, stream>>>(x, xb, NELEM);
  for (int p = 0; p < 4; ++p)
    k_f32_to_bf16<<<(D_MODEL * D_MODEL) / 256, 256, 0, stream>>>(wmat[p], wb[p],
                                                                 D_MODEL * D_MODEL);
  // 2) LoRA pre-products for q,k,v
  for (int p = 0; p < 3; ++p)
    k_xA_f32<<<(M_TOT * RANK) / 256, 256, 0, stream>>>(x, Amat[p], T[p]);

  // 3) projections
  dim3 ggrid(M_TOT / 64, D_MODEL / 256);
  // Q
  k_gemm_bf16<<<ggrid, 256, 0, stream>>>(xb, wb[0], Y0);
  k_epilogue_qkv<<<NELEM / 256, 256, 0, stream>>>(Y0, bias[0], T[0], Bmat[0], Qbuf, 0);
  // K
  k_gemm_bf16<<<ggrid, 256, 0, stream>>>(xb, wb[1], Y0);
  k_epilogue_qkv<<<NELEM / 256, 256, 0, stream>>>(Y0, bias[1], T[1], Bmat[1], Kbuf, 0);
  // V (transposed layout)
  k_gemm_bf16<<<ggrid, 256, 0, stream>>>(xb, wb[2], Y0);
  k_epilogue_qkv<<<NELEM / 256, 256, 0, stream>>>(Y0, bias[2], T[2], Bmat[2], Vtb, 1);

  // 4) attention
  k_attn_flash<<<dim3(SEQ / 64, N_HEADS, BATCH), 128, 0, stream>>>(
      Qbuf, Kbuf, Vtb, ctxb, rmax, rsum);
  k_attn_mean<<<dim3(SEQ / 16, SEQ / 128, BATCH), 256, 0, stream>>>(
      Qbuf, Kbuf, rmax, rsum, out + (size_t)BATCH * SEQ * SEQ);

  // 5) output projection + residual + LN
  k_xA_bf16<<<(M_TOT * RANK) / 256, 256, 0, stream>>>(ctxb, Amat[3], T[3]);
  k_gemm_bf16<<<ggrid, 256, 0, stream>>>(ctxb, wb[3], Y0);
  k_epilogue_o<<<NELEM / 256, 256, 0, stream>>>(Y0, bias[3], T[3], Bmat[3], x);
  k_layernorm<<<M_TOT, 256, 0, stream>>>(Y0, ln_g, ln_b, out);
}
